// NaryTreeLSTMCell_16604343566842
// MI455X (gfx1250) — matrix operand order repaired and checked by reading, not compiled, not run on hardware
//
#include <hip/hip_runtime.h>
#include <hip/hip_bf16.h>

typedef __attribute__((ext_vector_type(16))) _Float16 v16h;
typedef __attribute__((ext_vector_type(2)))  __fp16   v2fp;
typedef __attribute__((ext_vector_type(8)))  float    v8f;
typedef __attribute__((ext_vector_type(4)))  int      v4i;

#define WS_W_HALFS   (9 * 128 * 128)          // 147456 f16 weight fragments
#define WS_W_BYTES   (WS_W_HALFS * 2)         // 294912
#define WS_B_FLOATS  (5 * 128)                // combined biases
#define WS_BYTES     (WS_W_BYTES + WS_B_FLOATS * 4)   // 297472
#define SMEM_BYTES   WS_BYTES
#define SMEM_VEC16   (SMEM_BYTES / 16)        // 18592

#if defined(__gfx1250__) && __has_builtin(__builtin_amdgcn_global_load_async_to_lds_b128) && __has_builtin(__builtin_amdgcn_s_wait_asynccnt)
#define USE_ASYNC_LDS 1
#else
#define USE_ASYNC_LDS 0
#endif

struct PrepArgs {
    const float* W[9];   // Wi, Ui, Wf, Ufl, Ufr, Wo, Uo, Wu, Uu  (each [128,128], row=out)
    const float* B[9];   // matching biases [128]
};

// ---------------------------------------------------------------------------
// Prep: pack W^T into WMMA B-fragment layout (f16) + combine biases.
// B frag (32x16, f16): lane 0-15 -> col=lane, K=0..7,16..23 (2 halfs/VGPR);
//                      lane 16-31 -> col=lane-16, K=8..15,24..31.
// ws layout: [g][kt][nt][lane][16 halfs], frag = 1KB.
// ---------------------------------------------------------------------------
__global__ __launch_bounds__(256) void prep_weights(PrepArgs p, void* ws) {
    _Float16* wsW = (_Float16*)ws;
    float*    wsB = (float*)((char*)ws + WS_W_BYTES);
    int idx = blockIdx.x * 256 + threadIdx.x;
    if (idx < WS_W_HALFS) {
        int g  = idx >> 14;
        int r  = idx & 16383;
        int kt = r >> 12;
        int r2 = r & 4095;
        int nt = r2 >> 9;
        int r3 = r2 & 511;
        int ln = r3 >> 4;
        int h  = r3 & 15;
        int n  = nt * 16 + (ln & 15);
        int k  = kt * 32 + ((ln & 16) ? 8 : 0) + ((h & 8) ? 16 : 0) + (h & 7);
        wsW[idx] = (_Float16)p.W[g][n * 128 + k];
    } else if (idx < WS_W_HALFS + WS_B_FLOATS) {
        int j = idx - WS_W_HALFS;
        int which = j >> 7, c = j & 127;
        float v;
        switch (which) {
            case 0:  v = p.B[0][c] + p.B[1][c]; break;  // b_i  = Wi_b + Ui_b
            case 1:  v = p.B[2][c] + p.B[3][c]; break;  // b_fl = Wf_b + Ufl_b
            case 2:  v = p.B[2][c] + p.B[4][c]; break;  // b_fr = Wf_b + Ufr_b
            case 3:  v = p.B[5][c] + p.B[6][c]; break;  // b_o  = Wo_b + Uo_b
            default: v = p.B[7][c] + p.B[8][c]; break;  // b_u  = Wu_b + Uu_b
        }
        wsB[j] = v;
    }
}

// ---------------------------------------------------------------------------
__device__ __forceinline__ v16h pack16(float4 a, float4 b, float4 c, float4 d) {
    union { v16h v; v2fp h[8]; } u;
    u.h[0] = __builtin_amdgcn_cvt_pkrtz(a.x, a.y);
    u.h[1] = __builtin_amdgcn_cvt_pkrtz(a.z, a.w);
    u.h[2] = __builtin_amdgcn_cvt_pkrtz(b.x, b.y);
    u.h[3] = __builtin_amdgcn_cvt_pkrtz(b.z, b.w);
    u.h[4] = __builtin_amdgcn_cvt_pkrtz(c.x, c.y);
    u.h[5] = __builtin_amdgcn_cvt_pkrtz(c.z, c.w);
    u.h[6] = __builtin_amdgcn_cvt_pkrtz(d.x, d.y);
    u.h[7] = __builtin_amdgcn_cvt_pkrtz(d.z, d.w);
    return u.v;
}
__device__ __forceinline__ float4 f4add(float4 a, float4 b) {
    return make_float4(a.x + b.x, a.y + b.y, a.z + b.z, a.w + b.w);
}
__device__ __forceinline__ float sigm_(float x) { return 1.0f / (1.0f + __expf(-x)); }
__device__ __forceinline__ float tanh_(float x) { return 1.0f - 2.0f / (__expf(2.0f * x) + 1.0f); }

__device__ __forceinline__ void accum2(v8f& a0, v8f& a1, const v16h (&A)[2][4],
                                       int g, int nt, const _Float16* sW, int lane) {
#pragma unroll
    for (int kt = 0; kt < 4; ++kt) {
        v16h b = *(const v16h*)(sW + (size_t)(((g * 4 + kt) * 8 + nt) * 512 + lane * 16));
        a0 = __builtin_amdgcn_wmma_f32_16x16x32_f16(false, A[0][kt], false, b, (short)0, a0, false, false);
        a1 = __builtin_amdgcn_wmma_f32_16x16x32_f16(false, A[1][kt], false, b, (short)0, a1, false, false);
    }
}

// ---------------------------------------------------------------------------
// Main kernel: 8 waves/WG, 32 nodes/wave (two 16-row M-tiles).
// A frags (x, h_sum, h_l, h_r) held in VGPRs; B frags streamed from LDS.
// Weight staging via gfx1250 async global->LDS DMA (overlapped with A loads).
// ---------------------------------------------------------------------------
__global__ __launch_bounds__(256, 1)
void tree_lstm_wmma(const float* __restrict__ x,  const float* __restrict__ hl,
                    const float* __restrict__ cl, const float* __restrict__ hr,
                    const float* __restrict__ cr, const void* __restrict__ ws,
                    float* __restrict__ outH, float* __restrict__ outC, int nN) {
    extern __shared__ __align__(16) unsigned char smem[];
    _Float16* sW = (_Float16*)smem;
    float*    sB = (float*)(smem + WS_W_BYTES);

#if USE_ASYNC_LDS
    // Kick off the 297KB weight stage as async DMA first; it runs while we
    // issue and wait on the A-fragment global loads below.
    {
        __attribute__((address_space(1))) v4i* g =
            (__attribute__((address_space(1))) v4i*)ws;
        __attribute__((address_space(3))) v4i* l =
            (__attribute__((address_space(3))) v4i*)smem;
        for (int i = threadIdx.x; i < SMEM_VEC16; i += 256) {
            __builtin_amdgcn_global_load_async_to_lds_b128(g + i, l + i, 0, 0);
        }
    }
#endif

    const int lane = threadIdx.x & 31;
    const int wave = threadIdx.x >> 5;
    const int nodeBase = blockIdx.x * 256 + wave * 32;
    const int sel8 = (lane & 16) ? 8 : 0;
    const int rbase = nodeBase + (lane & 15);

    // ---- Load A fragments (16-bit A 16x32 layout), convert f32 -> f16 ----
    v16h Ax[2][4], Ahs[2][4], Ahl[2][4], Ahr[2][4];
#pragma unroll
    for (int t = 0; t < 2; ++t) {
        int row  = rbase + t * 16;
        int rowc = (row < nN) ? row : (nN - 1);
        const float* px = x  + (size_t)rowc * 128;
        const float* pl = hl + (size_t)rowc * 128;
        const float* pr = hr + (size_t)rowc * 128;
#pragma unroll
        for (int kt = 0; kt < 4; ++kt) {
            int c0 = kt * 32 + sel8;
            float4 x0 = *(const float4*)(px + c0);
            float4 x1 = *(const float4*)(px + c0 + 4);
            float4 x2 = *(const float4*)(px + c0 + 16);
            float4 x3 = *(const float4*)(px + c0 + 20);
            Ax[t][kt] = pack16(x0, x1, x2, x3);
            float4 l0 = *(const float4*)(pl + c0);
            float4 l1 = *(const float4*)(pl + c0 + 4);
            float4 l2 = *(const float4*)(pl + c0 + 16);
            float4 l3 = *(const float4*)(pl + c0 + 20);
            float4 r0 = *(const float4*)(pr + c0);
            float4 r1 = *(const float4*)(pr + c0 + 4);
            float4 r2 = *(const float4*)(pr + c0 + 16);
            float4 r3 = *(const float4*)(pr + c0 + 20);
            Ahl[t][kt] = pack16(l0, l1, l2, l3);
            Ahr[t][kt] = pack16(r0, r1, r2, r3);
            Ahs[t][kt] = pack16(f4add(l0, r0), f4add(l1, r1), f4add(l2, r2), f4add(l3, r3));
        }
    }

#if USE_ASYNC_LDS
    __builtin_amdgcn_s_wait_asynccnt(0);
#else
    // Fallback: staged copy through VGPRs (issued after the A loads so the
    // A-load latency hides under it).
    {
        const uint4* src = (const uint4*)ws;
        uint4*       dst = (uint4*)smem;
        for (int i = threadIdx.x; i < SMEM_VEC16; i += 256) dst[i] = src[i];
    }
#endif
    __syncthreads();

    const int colBase = lane & 15;

    // ---- Loop over 8 output n-tiles of 16 columns each ----
    for (int nt = 0; nt < 8; ++nt) {
        v8f i0 = {}, i1 = {}, f0 = {}, f1 = {}, o0 = {}, o1 = {}, u0 = {}, u1 = {};

        accum2(i0, i1, Ax,  0, nt, sW, lane);   // Wi  · x
        accum2(i0, i1, Ahs, 1, nt, sW, lane);   // Ui  · h_sum
        accum2(f0, f1, Ax,  2, nt, sW, lane);   // Wf  · x  (shared fx)
        v8f fl0 = f0, fl1 = f1, fr0 = f0, fr1 = f1;
        accum2(fl0, fl1, Ahl, 3, nt, sW, lane); // Ufl · h_left
        accum2(fr0, fr1, Ahr, 4, nt, sW, lane); // Ufr · h_right
        accum2(o0, o1, Ax,  5, nt, sW, lane);   // Wo  · x
        accum2(o0, o1, Ahs, 6, nt, sW, lane);   // Uo  · h_sum
        accum2(u0, u1, Ax,  7, nt, sW, lane);   // Wu  · x
        accum2(u0, u1, Ahs, 8, nt, sW, lane);   // Uu  · h_sum

        const int col = nt * 16 + colBase;
        const float bi  = sB[col];
        const float bfl = sB[128 + col];
        const float bfr = sB[256 + col];
        const float bo  = sB[384 + col];
        const float bu  = sB[512 + col];

#pragma unroll
        for (int t = 0; t < 2; ++t) {
            v8f ai  = t ? i1  : i0;
            v8f afl = t ? fl1 : fl0;
            v8f afr = t ? fr1 : fr0;
            v8f ao  = t ? o1  : o0;
            v8f au  = t ? u1  : u0;
#pragma unroll
            for (int r = 0; r < 8; ++r) {
                int row = nodeBase + t * 16 + ((lane & 16) ? r + 8 : r);
                if (row < nN) {
                    size_t off = (size_t)row * 128 + col;
                    float iv  = sigm_(ai[r]  + bi);
                    float flv = sigm_(afl[r] + bfl);
                    float frv = sigm_(afr[r] + bfr);
                    float ov  = sigm_(ao[r]  + bo);
                    float uv  = tanh_(au[r]  + bu);
                    float cv  = iv * uv + flv * cl[off] + frv * cr[off];
                    float hv  = ov * tanh_(cv);
                    outH[off] = hv;
                    outC[off] = cv;
                }
            }
        }
    }
}

// ---------------------------------------------------------------------------
extern "C" void kernel_launch(void* const* d_in, const int* in_sizes, int n_in,
                              void* d_out, int out_size, void* d_ws, size_t ws_size,
                              hipStream_t stream) {
    const float* x  = (const float*)d_in[0];
    const float* hl = (const float*)d_in[1];
    const float* cl = (const float*)d_in[2];
    const float* hr = (const float*)d_in[3];
    const float* cr = (const float*)d_in[4];

    PrepArgs pa;
    const int wIdx[9] = {5, 7, 9, 11, 13, 15, 17, 19, 21};  // Wi,Ui,Wf,Ufl,Ufr,Wo,Uo,Wu,Uu
    for (int g = 0; g < 9; ++g) {
        pa.W[g] = (const float*)d_in[wIdx[g]];
        pa.B[g] = (const float*)d_in[wIdx[g] + 1];
    }

    const int nN = in_sizes[0] / 128;
    float* outH = (float*)d_out;
    float* outC = outH + (size_t)nN * 128;

    // Pack weights/biases into workspace (tiny; same stream => ordered).
    {
        int total = WS_W_HALFS + WS_B_FLOATS;
        int blocks = (total + 255) / 256;
        prep_weights<<<blocks, 256, 0, stream>>>(pa, d_ws);
    }

    (void)hipFuncSetAttribute(reinterpret_cast<const void*>(tree_lstm_wmma),
                              hipFuncAttributeMaxDynamicSharedMemorySize, SMEM_BYTES);

    int nBlocks = (nN + 255) / 256;
    tree_lstm_wmma<<<nBlocks, 256, SMEM_BYTES, stream>>>(x, hl, cl, hr, cr, d_ws,
                                                         outH, outC, nN);
}
